// MultiModalMoE_16226386444687
// MI455X (gfx1250) — matrix operand order repaired and checked by reading, not compiled, not run on hardware
//
#include <hip/hip_runtime.h>
#include <hip/hip_bf16.h>
#include <math.h>

// Problem constants (B=16, S=1024 -> N=16384 tokens)
#define N_TOK  16384
#define DMODEL 256
#define NEXP   8
#define HDIM   1024
#define LN_EPSF 1e-5f

typedef __attribute__((ext_vector_type(16))) __bf16 v16bf;
typedef __attribute__((ext_vector_type(8)))  float  v8f;

// LDS row strides (elements) — padded vs bank conflicts, rows stay 16B-aligned
#define SA 272   // xe / W1-chunk rows (256 + 16)
#define SH 80    // h rows (64 + 16)
#define SW 72    // W2-chunk rows (64 + 8)

// ---------------------------------------------------------------------------
// CDNA5 async LDS-DMA via inline asm (GLOBAL_LOAD_ASYNC_TO_LDS_B128, ASYNCcnt).
// VDST = VGPR with LDS byte address, VADDR = 64-bit global address, saddr=off.
// Per ISA §10.2 the low 32 bits of a flat pointer to LDS are the LDS address,
// so a truncating cast of the generic pointer yields VDST directly.
__device__ __forceinline__ void async_copy_b128(const void* gsrc, void* ldst) {
  const unsigned lds_addr = (unsigned)(unsigned long long)ldst;
  asm volatile("global_load_async_to_lds_b128 %0, %1, off"
               :: "v"(lds_addr), "v"(gsrc)
               : "memory");
}

__device__ __forceinline__ void async_fence() {
#if __has_builtin(__builtin_amdgcn_s_wait_asynccnt)
  __builtin_amdgcn_s_wait_asynccnt(0);
#else
  asm volatile("s_wait_asynccnt 0x0" ::: "memory");
#endif
}

// ---------------------------------------------------------------------------
// WMMA fragment loaders (layouts per CDNA5 ISA §7.12.2, wave32).
// A (bf16 16x32): lane L holds row M = L&15; K runs {kbase+half*8 .. +7} in
// v0..v3 and {kbase+16+half*8 .. +7} in v4..v7 -> two contiguous 16B chunks.
__device__ __forceinline__ v16bf ldsA(const __bf16* base, int stride,
                                      int mbase, int kbase, int lane) {
  const __bf16* p = base + (mbase + (lane & 15)) * stride + kbase + ((lane >> 4) << 3);
  union { v16bf v; uint4 q[2]; } r;
  r.q[0] = *(const uint4*)(p);       // ds_load_b128
  r.q[1] = *(const uint4*)(p + 16);  // ds_load_b128
  return r.v;
}

// B (bf16 32x16) staged in [n][k] order: lane L holds col N = L&15, K runs
// {kbase+half*8 .. +7} then {kbase+16+half*8 .. +7} (same as A) -> 2x b128.
__device__ __forceinline__ v16bf ldsB(const __bf16* base, int stride,
                                      int nbase, int kbase, int lane) {
  const __bf16* p = base + (nbase + (lane & 15)) * stride + kbase + ((lane >> 4) << 3);
  union { v16bf v; uint4 q[2]; } r;
  r.q[0] = *(const uint4*)(p);
  r.q[1] = *(const uint4*)(p + 16);
  return r.v;
}

__device__ __forceinline__ float gelu_exact(float v) {
  return 0.5f * v * (1.0f + erff(v * 0.70710678118654752f));
}

// ---------------------------------------------------------------------------
__global__ __launch_bounds__(256) void zero_init_kernel(float* __restrict__ out,
                                                        int* __restrict__ counts) {
  const size_t i = (size_t)blockIdx.x * 256 + threadIdx.x;
  if (i < (size_t)N_TOK * DMODEL) out[i] = 0.0f;
  if (i < NEXP) counts[i] = 0;
}

// fp32 -> bf16 weight conversion with transpose into WMMA-B-friendly [n][k]:
//   W1  [E][D][H] -> W1t [E][H][D]   (fc1: n = h, k = d)
//   W2  [E][H][D] -> W2t [E][D][H]   (fc2: n = d, k = h)
__global__ __launch_bounds__(256) void convert_weights_kernel(
    const float* __restrict__ W1, const float* __restrict__ W2,
    __bf16* __restrict__ W1t, __bf16* __restrict__ W2t) {
  const size_t i = (size_t)blockIdx.x * 256 + threadIdx.x;
  const size_t EDH = (size_t)NEXP * DMODEL * HDIM;
  if (i >= EDH) return;
  const size_t e   = i / ((size_t)DMODEL * HDIM);
  const size_t rem = i % ((size_t)DMODEL * HDIM);
  { // W1 element [e][d][h]
    const size_t d = rem / HDIM, h = rem % HDIM;
    W1t[(e * HDIM + h) * DMODEL + d] = (__bf16)W1[i];
  }
  { // W2 element [e][h][d]
    const size_t h = rem / DMODEL, d = rem % DMODEL;
    W2t[(e * DMODEL + d) * HDIM + h] = (__bf16)W2[i];
  }
}

// ---------------------------------------------------------------------------
// Router + shared LayerNorm. One 256-thread block per token (tid == feature d).
__global__ __launch_bounds__(256) void router_ln_kernel(
    const float* __restrict__ x, const float* __restrict__ Wr,
    const float* __restrict__ br, __bf16* __restrict__ xn,
    int* __restrict__ counts, int* __restrict__ toklist,
    float* __restrict__ gates) {
  const int t = blockIdx.x, tid = threadIdx.x;
  __shared__ float red[256];
  const float xv = x[(size_t)t * DMODEL + tid];

  // mean
  red[tid] = xv; __syncthreads();
  for (int s = 128; s; s >>= 1) { if (tid < s) red[tid] += red[tid + s]; __syncthreads(); }
  const float mu = red[0] * (1.0f / DMODEL); __syncthreads();
  // variance
  const float dv = xv - mu;
  red[tid] = dv * dv; __syncthreads();
  for (int s = 128; s; s >>= 1) { if (tid < s) red[tid] += red[tid + s]; __syncthreads(); }
  const float var = red[0] * (1.0f / DMODEL); __syncthreads();
  xn[(size_t)t * DMODEL + tid] = (__bf16)(dv * rsqrtf(var + LN_EPSF));

  // router logits: 8 block reductions of x[d]*Wr[d][e]
  float lg[NEXP];
#pragma unroll
  for (int e = 0; e < NEXP; ++e) {
    red[tid] = xv * Wr[tid * NEXP + e]; __syncthreads();
    for (int s = 128; s; s >>= 1) { if (tid < s) red[tid] += red[tid + s]; __syncthreads(); }
    lg[e] = red[0] + br[e]; __syncthreads();
  }

  if (tid == 0) {
    // softmax is monotone -> top-2 on logits; renormalized gate = p0/(p0+p1)
    float mx = lg[0];
#pragma unroll
    for (int e = 1; e < NEXP; ++e) mx = fmaxf(mx, lg[e]);
    int i0 = 0;
#pragma unroll
    for (int e = 1; e < NEXP; ++e) if (lg[e] > lg[i0]) i0 = e;
    int i1 = (i0 == 0) ? 1 : 0;
#pragma unroll
    for (int e = 0; e < NEXP; ++e) if (e != i0 && lg[e] > lg[i1]) i1 = e;
    const float p0 = __expf(lg[i0] - mx), p1 = __expf(lg[i1] - mx);
    const float inv = 1.0f / (p0 + p1);
    int pos = atomicAdd(&counts[i0], 1);
    toklist[i0 * N_TOK + pos] = t; gates[i0 * N_TOK + pos] = p0 * inv;
    pos = atomicAdd(&counts[i1], 1);
    toklist[i1 * N_TOK + pos] = t; gates[i1 * N_TOK + pos] = p1 * inv;
  }
}

// ---------------------------------------------------------------------------
// Fused expert FFN: per block -> one expert e, one M-tile of 64 gathered tokens.
// fc1 (D=256 -> Hc=64 chunk) -> GELU -> fc2 (Hc -> D=256), accumulated over 16
// chunks, then gate-weighted atomic scatter into out.
__global__ __launch_bounds__(256) void moe_ffn_kernel(
    const __bf16* __restrict__ xn, const __bf16* __restrict__ W1t,
    const __bf16* __restrict__ W2t, const float* __restrict__ lng,
    const float* __restrict__ lnb, const float* __restrict__ b1,
    const float* __restrict__ b2, const int* __restrict__ counts,
    const int* __restrict__ toklist, const float* __restrict__ gates,
    float* __restrict__ out) {
  const int e    = blockIdx.x / (N_TOK / 64);
  const int tile = blockIdx.x % (N_TOK / 64);
  const int cnt  = counts[e];
  const int m0   = tile * 64;
  if (m0 >= cnt) return;                       // empty tile for this expert
  const int rows = min(64, cnt - m0);

  extern __shared__ __bf16 smem[];
  __bf16* xeL = smem;                 // [64][SA]  affine-normalized activations
  __bf16* w1L = xeL + 64 * SA;        // [64][SA]  W1t chunk (n=h_local, k=d)
  __bf16* hL  = w1L + 64 * SA;        // [64][SH]  GELU(fc1) chunk (m, h_local)
  __bf16* w2L = hL  + 64 * SH;        // [256][SW] W2t chunk (n=d, k=h_local)

  const int tid = threadIdx.x, wv = tid >> 5, lane = tid & 31;
  const int r  = wv & 3;   // M row-tile (16 rows) owned by this wave
  const int cg = wv >> 2;  // column group: y col-tiles cg*8..cg*8+7

  // ---- stage xe = xn * ln_g[e] + ln_b[e] (zero-pad rows >= count) ----
  for (int idx = tid; idx < 64 * DMODEL; idx += 256) {
    const int m = idx >> 8, d = idx & 255;
    float v = 0.0f;
    if (m < rows) {
      const int t = toklist[e * N_TOK + m0 + m];
      v = (float)xn[(size_t)t * DMODEL + d] * lng[e * DMODEL + d] + lnb[e * DMODEL + d];
    }
    xeL[m * SA + d] = (__bf16)v;
  }

  v8f yacc[8];
#pragma unroll
  for (int j = 0; j < 8; ++j) {
    v8f z = {0.f, 0.f, 0.f, 0.f, 0.f, 0.f, 0.f, 0.f};
    yacc[j] = z;
  }

  for (int hc = 0; hc < HDIM; hc += 64) {
    __syncthreads();  // xe ready (iter 0) / previous fc2 done with w1L,w2L,hL

    // ---- stage W1t chunk (async DMA global -> LDS): 64 rows, 256 k each ----
    for (int idx = tid; idx < (64 * 256) / 8; idx += 256) {
      const int n = idx >> 5, k8 = (idx & 31) << 3;
      async_copy_b128(W1t + ((size_t)(e * HDIM + hc + n)) * DMODEL + k8,
                      w1L + n * SA + k8);
    }
    // ---- stage W2t chunk (async DMA): 256 rows (n = d), 64 k each ----
    for (int idx = tid; idx < (256 * 64) / 8; idx += 256) {
      const int n = idx >> 3, k8 = (idx & 7) << 3;
      async_copy_b128(W2t + ((size_t)(e * DMODEL + n)) * HDIM + hc + k8,
                      w2L + n * SW + k8);
    }
    // hint the next chunk into cache (global_prefetch_b8)
    if (hc + 64 < HDIM)
      __builtin_prefetch(W1t + ((size_t)(e * HDIM + hc + 64 + (tid & 63))) * DMODEL, 0, 1);
    async_fence();                               // s_wait_asynccnt 0
    __syncthreads();

    // ---- fc1: h[64][64] chunk; wave owns row-tile r, col-tiles cg*2, cg*2+1 ----
    v8f h0 = {0.f,0.f,0.f,0.f,0.f,0.f,0.f,0.f};
    v8f h1 = {0.f,0.f,0.f,0.f,0.f,0.f,0.f,0.f};
#pragma unroll
    for (int k = 0; k < DMODEL; k += 32) {
      const v16bf a  = ldsA(xeL, SA, r * 16, k, lane);
      const v16bf bA = ldsB(w1L, SA, (cg * 2 + 0) * 16, k, lane);
      const v16bf bB = ldsB(w1L, SA, (cg * 2 + 1) * 16, k, lane);
      h0 = __builtin_amdgcn_wmma_f32_16x16x32_bf16(false, a, false, bA, (short)0, h0, false, false);
      h1 = __builtin_amdgcn_wmma_f32_16x16x32_bf16(false, a, false, bB, (short)0, h1, false, false);
    }

    // ---- bias + exact GELU, write h chunk to LDS (C-layout: col=lane&15, row=v+8*half) ----
    {
      const int n0 = (cg * 2 + 0) * 16 + (lane & 15);
      const int n1 = (cg * 2 + 1) * 16 + (lane & 15);
      const float bb0 = b1[e * HDIM + hc + n0];
      const float bb1 = b1[e * HDIM + hc + n1];
#pragma unroll
      for (int v = 0; v < 8; ++v) {
        const int mrow = r * 16 + v + ((lane >> 4) << 3);
        hL[mrow * SH + n0] = (__bf16)gelu_exact(h0[v] + bb0);
        hL[mrow * SH + n1] = (__bf16)gelu_exact(h1[v] + bb1);
      }
    }
    __syncthreads();

    // ---- fc2: y[64][256] += h[64][64] @ W2chunk[64][256] ----
    const v16bf a0 = ldsA(hL, SH, r * 16, 0, lane);
    const v16bf a1 = ldsA(hL, SH, r * 16, 32, lane);
#pragma unroll
    for (int j = 0; j < 8; ++j) {
      const int ct = cg * 8 + j;
      const v16bf bA = ldsB(w2L, SW, ct * 16, 0, lane);
      const v16bf bB = ldsB(w2L, SW, ct * 16, 32, lane);
      yacc[j] = __builtin_amdgcn_wmma_f32_16x16x32_bf16(false, a0, false, bA, (short)0, yacc[j], false, false);
      yacc[j] = __builtin_amdgcn_wmma_f32_16x16x32_bf16(false, a1, false, bB, (short)0, yacc[j], false, false);
    }
  }

  // ---- gate-weighted scatter: out[t] += g * (y + b2[e]) ----
#pragma unroll
  for (int v = 0; v < 8; ++v) {
    const int mrow = r * 16 + v + ((lane >> 4) << 3);
    if (mrow < rows) {
      const int   t = toklist[e * N_TOK + m0 + mrow];
      const float g = gates[e * N_TOK + m0 + mrow];
#pragma unroll
      for (int j = 0; j < 8; ++j) {
        const int col = (cg * 8 + j) * 16 + (lane & 15);
        atomicAdd(out + (size_t)t * DMODEL + col,
                  g * (yacc[j][v] + b2[e * DMODEL + col]));
      }
    }
  }
}

// ---------------------------------------------------------------------------
extern "C" void kernel_launch(void* const* d_in, const int* in_sizes, int n_in,
                              void* d_out, int out_size, void* d_ws, size_t ws_size,
                              hipStream_t stream) {
  const float* x   = (const float*)d_in[0];
  const float* Wr  = (const float*)d_in[1];
  const float* br  = (const float*)d_in[2];
  const float* lng = (const float*)d_in[3];
  const float* lnb = (const float*)d_in[4];
  const float* W1  = (const float*)d_in[5];
  const float* b1  = (const float*)d_in[6];
  const float* W2  = (const float*)d_in[7];
  const float* b2  = (const float*)d_in[8];
  float* out = (float*)d_out;

  // workspace carve (~17.1 MB)
  char* ws = (char*)d_ws;
  __bf16* xn   = (__bf16*)(ws);                            // 8 MB
  __bf16* W1t  = (__bf16*)(ws + (size_t)8  * 1024 * 1024); // 4 MB
  __bf16* W2t  = (__bf16*)(ws + (size_t)12 * 1024 * 1024); // 4 MB
  int*    cnts = (int*)   (ws + (size_t)16 * 1024 * 1024);
  int*    tokl = (int*)   (ws + (size_t)16 * 1024 * 1024 + 256);
  float*  gate = (float*) (ws + (size_t)16 * 1024 * 1024 + 256 +
                           (size_t)NEXP * N_TOK * 4);

  zero_init_kernel<<<(N_TOK * DMODEL) / 256, 256, 0, stream>>>(out, cnts);
  convert_weights_kernel<<<(NEXP * DMODEL * HDIM) / 256, 256, 0, stream>>>(W1, W2, W1t, W2t);
  router_ln_kernel<<<N_TOK, 256, 0, stream>>>(x, Wr, br, xn, cnts, tokl, gate);

  const size_t lds_bytes = (size_t)(64 * SA + 64 * SA + 64 * SH + 256 * SW) * sizeof(__bf16);
  moe_ffn_kernel<<<NEXP * (N_TOK / 64), 256, lds_bytes, stream>>>(
      xn, W1t, W2t, lng, lnb, b1, b2, cnts, tokl, gate, out);
}
